// YOLOLayer_15401752723829
// MI455X (gfx1250) — matrix operand order repaired and checked by reading, not compiled, not run on hardware
//
#include <hip/hip_runtime.h>
#include <hip/hip_bf16.h>

#define GG   5776      // 76*76
#define NCH  255       // 3*85 output channels
#define CIN  256
#define NB   16
#define NA   3
#define NGT  50

typedef _Float16 v16h __attribute__((ext_vector_type(16)));
typedef _Float16 v8h  __attribute__((ext_vector_type(8)));
typedef float    v8f  __attribute__((ext_vector_type(8)));

__device__ __forceinline__ float sigmoidf(float x) { return 1.0f / (1.0f + __expf(-x)); }
__device__ __forceinline__ float clogf(float x)    { return fmaxf(__logf(x), -100.0f); }
__device__ __forceinline__ float bcef(float p, float t) {
    return -(t * clogf(p) + (1.0f - t) * clogf(1.0f - p));
}

// ---------------------------------------------------------------- init
__global__ void init_kernel(int* __restrict__ map, float* __restrict__ out, int n) {
    int i = blockIdx.x * blockDim.x + threadIdx.x;
    if (i < n) map[i] = -1;
    if (blockIdx.x == 0 && threadIdx.x < 5) out[threadIdx.x] = 0.0f;
}

// ---------------------------------------------------------------- gt -> target records
__global__ void match_kernel(const float* __restrict__ labels,
                             int* __restrict__ map, float* __restrict__ recs) {
    int idx = blockIdx.x * blockDim.x + threadIdx.x;
    if (idx >= NB * NGT) return;
    int b = idx / NGT;
    const float* l = labels + (size_t)idx * 5;
    float s = l[0] + l[1] + l[2] + l[3] + l[4];
    bool valid = s > 0.0f;
    float gx = l[1] * 76.0f, gy = l[2] * 76.0f, gw = l[3] * 76.0f, gh = l[4] * 76.0f;
    const float AW[9] = {1.25f, 2.0f, 4.125f, 3.75f, 7.75f, 7.375f, 14.5f, 19.5f, 46.625f};
    const float AH[9] = {1.625f, 3.75f, 2.875f, 7.625f, 5.625f, 14.875f, 11.25f, 24.75f, 40.75f};
    float area_g = gw * gh;
    int best = 0; float bi = -1.0f;
    #pragma unroll
    for (int k = 0; k < 9; ++k) {
        float inter = fminf(gw, AW[k]) * fminf(gh, AH[k]);
        float iou = inter / (area_g + AW[k] * AH[k] - inter);
        if (iou > bi) { bi = iou; best = k; }          // first max wins (strict >)
    }
    int a = (valid && best < NA) ? best : -1;
    if (a < 0) return;
    float fx = floorf(gx), fy = floorf(gy);
    int gi = (int)fx, gj = (int)fy;
    gi = gi < 0 ? 0 : (gi > 75 ? 75 : gi);
    gj = gj < 0 ? 0 : (gj > 75 ? 75 : gj);
    float* rec = recs + (size_t)idx * 86;
    for (int k = 0; k < 85; ++k) rec[k] = 0.0f;
    rec[0] = gx - fx;
    rec[1] = gy - fy;
    rec[2] = __logf(gw / AW[a] + 1e-16f);
    rec[3] = __logf(gh / AH[a] + 1e-16f);
    rec[4] = 1.0f;
    int cls = (int)l[0];
    if (cls >= 0 && cls < 80) rec[5 + cls] = 1.0f;
    rec[85] = 2.0f - gw * gh / (76.0f * 76.0f);
    int cell = ((b * NA + a) * 76 + gj) * 76 + gi;
    map[cell] = idx;
}

// ---------------------------------------------------------------- 1x1 conv = GEMM via WMMA f16
// preds[b, o, p] = sum_c W[o,c] * X[b,c,p]   (bias added in consumer)
// Block: 128 output rows (8 M-tiles, W staged in LDS as f16) x 128 pixels (8 waves,
// one 16-col N-tile each). Each wave keeps 8 accumulator tiles in VGPRs and reuses
// one B fragment across 8 WMMAs per K-chunk -> X is read only 2x total.
__global__ __launch_bounds__(256) void conv_gemm(const float* __restrict__ xin,
                                                 const float* __restrict__ conv_w,
                                                 float* __restrict__ preds) {
    __shared__ _Float16 ldsW[128 * 264];         // 128 rows x 256 K (stride 264: conflict-free)
    const int b  = blockIdx.z;
    const int m0 = blockIdx.y * 128;
    const int t  = threadIdx.x;

    // cooperative W stage + f32->f16 convert (rows past 254 zero-padded)
    for (int r = 0; r < 128; ++r) {
        int row = m0 + r;
        float w = (row < NCH) ? conv_w[(size_t)row * CIN + t] : 0.0f;
        ldsW[r * 264 + t] = (_Float16)w;
    }
    __syncthreads();

    const int warp  = t >> 5;
    const int lane  = t & 31;
    const int tileN = blockIdx.x * 8 + warp;     // 361 N-tiles total (5776/16)
    if (tileN >= 361) return;                    // wave-uniform skip
    const int n0   = tileN * 16;
    const int half = lane >> 4;                  // lane group 0/1
    const int l16  = lane & 15;
    const int akb  = half * 8;                   // A K-base within chunk of 16
    const int bkb  = half * 16;                  // K offset of prefetched row in next chunk of 32
    const float* xb = xin + (size_t)b * CIN * GG + (n0 + l16);

    v8f acc[8] = {};
    for (int kc = 0; kc < CIN; kc += 32) {
        // B fragment: col N=l16, K = kc+akb+2i (+8 for i>=4) / +1, matching A layout.
        v16h bf;
        #pragma unroll
        for (int i = 0; i < 8; ++i) {
            int k = kc + akb + 2 * i + (i >= 4 ? 8 : 0);
            float x0 = xb[(size_t)k * GG];
            float x1 = xb[(size_t)(k + 1) * GG];
            bf[2 * i]     = (_Float16)x0;
            bf[2 * i + 1] = (_Float16)x1;
        }
        if (kc + 32 < CIN)
            __builtin_prefetch(&xb[(size_t)(kc + 32 + bkb) * GG], 0, 3);

        // 8 M-tiles reuse this B fragment; A from LDS per ISA 16-bit A layout.
        #pragma unroll
        for (int mt = 0; mt < 8; ++mt) {
            const _Float16* arow = &ldsW[(mt * 16 + l16) * 264 + kc + akb];
            v8h alo = *(const v8h*)(arow);       // K = kc+akb   .. +7
            v8h ahi = *(const v8h*)(arow + 16);  // K = kc+16+akb .. +7
            v16h a;
            #pragma unroll
            for (int u = 0; u < 8; ++u) { a[u] = alo[u]; a[8 + u] = ahi[u]; }
            acc[mt] = __builtin_amdgcn_wmma_f32_16x16x32_f16(false, a, false, bf,
                                                             (short)0, acc[mt],
                                                             false, false);
        }
    }

    // D store per ISA 32-bit C/D layout: VGPR r -> M = r + half*8, N = l16
    const int col = n0 + l16;
    #pragma unroll
    for (int mt = 0; mt < 8; ++mt) {
        #pragma unroll
        for (int r = 0; r < 8; ++r) {
            int o = m0 + mt * 16 + r + half * 8;
            if (o < NCH)
                preds[((size_t)b * NCH + o) * GG + col] = acc[mt][r];
        }
    }
}

// ---------------------------------------------------------------- per-cell loss + target_out
__global__ __launch_bounds__(256) void loss_kernel(const float* __restrict__ preds,
                                                   const float* __restrict__ conv_b,
                                                   const float* __restrict__ labels,
                                                   const int* __restrict__ map,
                                                   const float* __restrict__ recs,
                                                   float* __restrict__ out) {
    __shared__ float gbox[NGT][4];
    __shared__ float gval[NGT];
    __shared__ float sbias[NCH];
    __shared__ float4 red[256];

    const int b = blockIdx.y;
    const int t = threadIdx.x;
    if (t < NGT) {
        const float* l = labels + ((size_t)b * NGT + t) * 5;
        float s = l[0] + l[1] + l[2] + l[3] + l[4];
        gval[t] = (s > 0.0f) ? 1.0f : 0.0f;
        gbox[t][0] = l[1] * 76.0f; gbox[t][1] = l[2] * 76.0f;
        gbox[t][2] = l[3] * 76.0f; gbox[t][3] = l[4] * 76.0f;
    }
    if (t < NCH) sbias[t] = conv_b[t];
    __syncthreads();

    const float AWL[NA] = {1.25f, 2.0f, 4.125f};
    const float AHL[NA] = {1.625f, 3.75f, 2.875f};

    int idx = blockIdx.x * 256 + t;
    float lxy = 0.0f, lwh = 0.0f, lobj = 0.0f, lcls = 0.0f;
    if (idx < NA * GG) {
        int a = idx / GG;
        int p = idx - a * GG;
        int j = p / 76, i = p - j * 76;
        const float* pc = preds + ((size_t)b * NCH + a * 85) * GG + p;
        float p0 = pc[0]              + sbias[a * 85 + 0];
        float p1 = pc[(size_t)1 * GG] + sbias[a * 85 + 1];
        float p2 = pc[(size_t)2 * GG] + sbias[a * 85 + 2];
        float p3 = pc[(size_t)3 * GG] + sbias[a * 85 + 3];
        float p4 = pc[(size_t)4 * GG] + sbias[a * 85 + 4];

        float sx = sigmoidf(p0), sy = sigmoidf(p1);
        float bx = sx + (float)i, by = sy + (float)j;
        float bw = __expf(p2) * AWL[a], bh = __expf(p3) * AHL[a];
        float areaA = bw * bh;
        float ax0 = bx - 0.5f * bw, ay0 = by - 0.5f * bh;
        float ax1 = bx + 0.5f * bw, ay1 = by + 0.5f * bh;

        float maxiou = 0.0f;
        #pragma unroll 10
        for (int g = 0; g < NGT; ++g) {
            float gx = gbox[g][0], gy = gbox[g][1], gw = gbox[g][2], gh = gbox[g][3];
            float tlx = fmaxf(ax0, gx - 0.5f * gw), tly = fmaxf(ay0, gy - 0.5f * gh);
            float brx = fminf(ax1, gx + 0.5f * gw), bry = fminf(ay1, gy + 0.5f * gh);
            float inter = (tlx < brx && tly < bry) ? (brx - tlx) * (bry - tly) : 0.0f;
            float iou = inter / (areaA + gw * gh - inter);
            maxiou = fmaxf(maxiou, iou * gval[g]);
        }
        float objm0 = (maxiou <= 0.7f) ? 1.0f : 0.0f;

        int cell = ((b * NA + a) * 76 + j) * 76 + i;
        int r = map[cell];
        float* to = out + 5 + (size_t)cell * 85;
        float so = sigmoidf(p4);
        if (r >= 0) {
            const float* rec = recs + (size_t)r * 86;
            float sc = rec[85];
            lxy = (bcef(sx, rec[0]) + bcef(sy, rec[1])) * sc;
            float dw = p2 - rec[2], dh = p3 - rec[3];
            lwh = 0.5f * sc * (dw * dw + dh * dh);
            lobj = bcef(so, 1.0f);                      // obj_mask forced 1 at target
            to[0] = rec[0]; to[1] = rec[1]; to[2] = rec[2]; to[3] = rec[3]; to[4] = 1.0f;
            for (int cc = 0; cc < 80; ++cc) {
                float pv = sigmoidf(pc[(size_t)(5 + cc) * GG] + sbias[a * 85 + 5 + cc]);
                lcls += bcef(pv, rec[5 + cc]);
                to[5 + cc] = rec[5 + cc];
            }
        } else {
            lobj = objm0 * (-clogf(1.0f - so));         // BCE(so, 0) where obj_mask
            for (int k = 0; k < 85; ++k) to[k] = 0.0f;
        }
    }

    red[t] = make_float4(lxy, lwh, lobj, lcls);
    __syncthreads();
    for (int s2 = 128; s2 > 0; s2 >>= 1) {
        if (t < s2) {
            red[t].x += red[t + s2].x; red[t].y += red[t + s2].y;
            red[t].z += red[t + s2].z; red[t].w += red[t + s2].w;
        }
        __syncthreads();
    }
    if (t == 0) {
        atomicAdd(&out[1], red[0].x);
        atomicAdd(&out[2], red[0].y);
        atomicAdd(&out[3], red[0].z);
        atomicAdd(&out[4], red[0].w);
        atomicAdd(&out[0], red[0].x + red[0].y + red[0].z + red[0].w);
    }
}

// ---------------------------------------------------------------- launcher
extern "C" void kernel_launch(void* const* d_in, const int* in_sizes, int n_in,
                              void* d_out, int out_size, void* d_ws, size_t ws_size,
                              hipStream_t stream) {
    const float* xin    = (const float*)d_in[0];   // [16,256,76,76]
    const float* labels = (const float*)d_in[1];   // [16,50,5]
    const float* conv_w = (const float*)d_in[2];   // [255,256]
    const float* conv_b = (const float*)d_in[3];   // [255]
    float* out = (float*)d_out;                    // 5 losses + target_out [16,3,76,76,85]

    float* preds = (float*)d_ws;                                   // 16*255*5776 f32
    int*   map   = (int*)(preds + (size_t)NB * NCH * GG);          // 16*3*5776 i32
    float* recs  = (float*)(map + (size_t)NB * NA * GG);           // 800*86 f32

    const int ncell = NB * NA * GG;
    init_kernel<<<(ncell + 255) / 256, 256, 0, stream>>>(map, out, ncell);
    match_kernel<<<(NB * NGT + 255) / 256, 256, 0, stream>>>(labels, map, recs);
    conv_gemm<<<dim3(46, 2, NB), 256, 0, stream>>>(xin, conv_w, preds);
    loss_kernel<<<dim3(68, NB), 256, 0, stream>>>(preds, conv_b, labels, map, recs, out);
}
